// geometric_attention_71665824301378
// MI455X (gfx1250) — matrix operand order repaired and checked by reading, not compiled, not run on hardware
//
#include <hip/hip_runtime.h>
#include <math.h>

#define HEADS 8
#define NSEQ  4096
#define DPAD  320     // padded q/k feature dim: 10 x K32 WMMA steps
#define NQK   304     // real q/k feature dim (n_ch in reference)
#define DVROWS 272    // real v dim: 17 x 16 WMMA N-tiles
#define MV_OUT ((size_t)HEADS * NSEQ * 256)

typedef __bf16 bf16;
typedef __bf16 v16bf __attribute__((ext_vector_type(16)));
typedef __bf16 v8bf  __attribute__((ext_vector_type(8)));
typedef float  v8f   __attribute__((ext_vector_type(8)));

// ---------------------------------------------------------------------------
// Kernel 1: feature construction.
//   Q/K rows: [240 ip | 48 dist | 16 scalar | 16 zero-pad] in bf16, row-major.
//   Score scale 1/sqrt(304) folded into Q; k-rescale sqrt(n_ch/n_qk) into K.
//   V stored transposed (d-major, n contiguous) so the PV WMMA B-operand is a
//   single contiguous 16-bf16 load per lane.
// ---------------------------------------------------------------------------
__global__ __launch_bounds__(256) void ga_prep(
    const float* __restrict__ q_mv, const float* __restrict__ k_mv,
    const float* __restrict__ v_mv, const float* __restrict__ q_s,
    const float* __restrict__ k_s,  const float* __restrict__ v_s,
    const float* __restrict__ basis_q, const float* __restrict__ basis_k,
    bf16* __restrict__ Q, bf16* __restrict__ Kf, bf16* __restrict__ Vt)
{
  int idx = blockIdx.x * blockDim.x + threadIdx.x;   // h*NSEQ + n
  if (idx >= HEADS * NSEQ) return;
  int h = idx / NSEQ;
  int n = idx - h * NSEQ;

  const float qscale = rsqrtf(304.0f);               // 1/sqrt(n_ch)
  const float kscale = sqrtf(304.0f / 304.0f);       // sqrt(n_ch/n_qk) == 1

  for (int which = 0; which < 2; ++which) {
    const float* mv = (which ? k_mv : q_mv) + (size_t)idx * 8 * 32;
    const float* sc = (which ? k_s  : q_s ) + (size_t)idx * 16;
    const float* bs = which ? basis_k : basis_q;
    const float  fs = which ? kscale : qscale;
    bf16* row = (which ? Kf : Q) + (size_t)idx * DPAD;

    for (int c = 0; c < 8; ++c) {
      const float* m = mv + c * 32;
      #pragma unroll
      for (int j = 0; j < 30; ++j)
        row[c * 30 + j] = (bf16)(m[1 + j] * fs);

      float t[5];
      float nr = rsqrtf(m[5] * m[5] + 1.0e-3f);      // normalizer(tri[...,4])
      #pragma unroll
      for (int x = 0; x < 5; ++x) t[x] = m[1 + x] * nr;
      #pragma unroll
      for (int z = 0; z < 6; ++z) {
        float acc = 0.f;
        #pragma unroll
        for (int x = 0; x < 5; ++x)
          #pragma unroll
          for (int y = 0; y < 5; ++y)
            acc += bs[(x * 5 + y) * 6 + z] * t[x] * t[y];
        row[240 + c * 6 + z] = (bf16)(acc * fs);
      }
    }
    for (int s = 0; s < 16; ++s) row[288 + s] = (bf16)(sc[s] * fs);
    for (int d = NQK; d < DPAD; ++d) row[d] = (bf16)0.f;
  }

  // transposed V: Vt[h][d][n], n contiguous -> coalesced across threads
  bf16* vth = Vt + (size_t)h * DVROWS * NSEQ;
  const float* vm = v_mv + (size_t)idx * 256;
  for (int d = 0; d < 256; ++d) vth[(size_t)d * NSEQ + n] = (bf16)vm[d];
  const float* vs = v_s + (size_t)idx * 16;
  for (int s = 0; s < 16; ++s)  vth[(size_t)(256 + s) * NSEQ + n] = (bf16)vs[s];
}

// ---------------------------------------------------------------------------
// Kernel 2: flash attention. 4 waves/block, 16 query rows per wave.
// Per 32-key tile: 2x10 WMMA (scores, d=320) + online softmax (shfl_xor row
// reductions within 16-lane halves, matching the C/D tile layout) + LDS bounce
// to rebuild P in A-fragment layout + 17 WMMA (P x V^T).
// ---------------------------------------------------------------------------
__global__ __launch_bounds__(128) void ga_attn(
    const bf16* __restrict__ Q, const bf16* __restrict__ Kf,
    const bf16* __restrict__ Vt, float* __restrict__ out)
{
  __shared__ bf16 lds_p[4][16 * 32];                 // per-wave 16x32 P tile

  const int lane  = threadIdx.x & 31;
  const int wave  = threadIdx.x >> 5;
  const int h     = blockIdx.x >> 6;                 // 64 blocks per head
  const int qblk  = blockIdx.x & 63;
  const int qbase = qblk * 64 + wave * 16;
  const int row   = lane & 15;
  const int hiH   = lane >> 4;                       // 0: lanes 0-15, 1: 16-31

  // Preload Q A-fragments: lane holds row (lane&15); low half-wave loads
  // K {0..15}, high half-wave {16..31} of each K32 step.
  const bf16* qrow = Q + (size_t)(h * NSEQ + qbase + row) * DPAD + hiH * 16;
  v16bf aq[10];
  #pragma unroll
  for (int dt = 0; dt < 10; ++dt) {
    aq[dt] = *(const v16bf*)(qrow + dt * 32);
  }

  const bf16* kbase = Kf + (size_t)h * NSEQ * DPAD + hiH * 16;
  const bf16* vbase = Vt + (size_t)h * DVROWS * NSEQ + (size_t)row * NSEQ + hiH * 16;

  float m[8], l[8];
  v8f acc[17];
  #pragma unroll
  for (int r = 0; r < 8; ++r) { m[r] = -3.0e38f; l[r] = 0.f; }
  #pragma unroll
  for (int t = 0; t < 17; ++t)
    #pragma unroll
    for (int r = 0; r < 8; ++r) acc[t][r] = 0.f;

  for (int kb = 0; kb < NSEQ; kb += 32) {
    // ---- scores: S = Q x K^T over d=320 (10 x K32), two 16-key subtiles ----
    v8f s0, s1;
    #pragma unroll
    for (int r = 0; r < 8; ++r) { s0[r] = 0.f; s1[r] = 0.f; }
    const bf16* kp0 = kbase + (size_t)(kb + row) * DPAD;
    const bf16* kp1 = kbase + (size_t)(kb + 16 + row) * DPAD;
    #pragma unroll
    for (int dt = 0; dt < 10; ++dt) {
      v16bf b0 = *(const v16bf*)(kp0 + dt * 32);
      v16bf b1 = *(const v16bf*)(kp1 + dt * 32);
      s0 = __builtin_amdgcn_wmma_f32_16x16x32_bf16(false, aq[dt], false, b0,
                                                   (short)0, s0, false, false);
      s1 = __builtin_amdgcn_wmma_f32_16x16x32_bf16(false, aq[dt], false, b1,
                                                   (short)0, s1, false, false);
    }

    // ---- online softmax; row r of lane group == accumulator VGPR r ----
    float alpha[8];
    #pragma unroll
    for (int r = 0; r < 8; ++r) {
      float tm = fmaxf(s0[r], s1[r]);
      tm = fmaxf(tm, __shfl_xor(tm, 1, 32));
      tm = fmaxf(tm, __shfl_xor(tm, 2, 32));
      tm = fmaxf(tm, __shfl_xor(tm, 4, 32));
      tm = fmaxf(tm, __shfl_xor(tm, 8, 32));
      float nm = fmaxf(m[r], tm);
      alpha[r] = __expf(m[r] - nm);
      m[r] = nm;
      float p0 = __expf(s0[r] - nm);
      float p1 = __expf(s1[r] - nm);
      s0[r] = p0; s1[r] = p1;
      float rs = p0 + p1;
      rs += __shfl_xor(rs, 1, 32);
      rs += __shfl_xor(rs, 2, 32);
      rs += __shfl_xor(rs, 4, 32);
      rs += __shfl_xor(rs, 8, 32);
      l[r] = l[r] * alpha[r] + rs;
    }
    #pragma unroll
    for (int t = 0; t < 17; ++t)
      #pragma unroll
      for (int r = 0; r < 8; ++r) acc[t][r] *= alpha[r];

    // ---- P tile -> LDS (C/D layout) -> reread as A fragment ----
    bf16* lp = lds_p[wave];
    #pragma unroll
    for (int r = 0; r < 8; ++r) {
      int mr = r + hiH * 8;
      lp[mr * 32 + row]      = (bf16)s0[r];
      lp[mr * 32 + 16 + row] = (bf16)s1[r];
    }
    asm volatile("s_wait_dscnt 0" ::: "memory");     // per-wave LDS reuse
    const bf16* ap = lp + row * 32 + hiH * 16;
    v16bf pa = *(const v16bf*)ap;

    // ---- O += P x V (V transposed: contiguous 32B per lane) ----
    #pragma unroll
    for (int t = 0; t < 17; ++t) {
      v16bf bv = *(const v16bf*)(vbase + (size_t)t * 16 * NSEQ + kb);
      acc[t] = __builtin_amdgcn_wmma_f32_16x16x32_bf16(false, pa, false, bv,
                                                       (short)0, acc[t], false, false);
    }
  }

  // ---- epilogue: normalize, scatter to (v_out_mv | v_out_s) ----
  #pragma unroll
  for (int r = 0; r < 8; ++r) l[r] = 1.0f / l[r];
  const int q0 = qbase + hiH * 8;
  #pragma unroll
  for (int t = 0; t < 17; ++t) {
    int dv = t * 16 + row;
    #pragma unroll
    for (int r = 0; r < 8; ++r) {
      float val = acc[t][r] * l[r];
      int qq = q0 + r;
      size_t o = (dv < 256)
          ? ((size_t)(h * NSEQ + qq)) * 256 + dv
          : MV_OUT + ((size_t)(h * NSEQ + qq)) * 16 + (size_t)(dv - 256);
      out[o] = val;
    }
  }
}

// ---------------------------------------------------------------------------
extern "C" void kernel_launch(void* const* d_in, const int* in_sizes, int n_in,
                              void* d_out, int out_size, void* d_ws, size_t ws_size,
                              hipStream_t stream) {
  (void)in_sizes; (void)n_in; (void)out_size; (void)ws_size;
  const float* q_mv    = (const float*)d_in[0];
  const float* k_mv    = (const float*)d_in[1];
  const float* v_mv    = (const float*)d_in[2];
  const float* q_s     = (const float*)d_in[3];
  const float* k_s     = (const float*)d_in[4];
  const float* v_s     = (const float*)d_in[5];
  const float* basis_q = (const float*)d_in[6];
  const float* basis_k = (const float*)d_in[7];

  // workspace layout (bf16): Q[8*4096*320] | K[8*4096*320] | Vt[8*272*4096]
  bf16* Q  = (bf16*)d_ws;
  bf16* Kf = Q  + (size_t)HEADS * NSEQ * DPAD;
  bf16* Vt = Kf + (size_t)HEADS * NSEQ * DPAD;       // ~59.8 MB total

  ga_prep<<<(HEADS * NSEQ + 255) / 256, 256, 0, stream>>>(
      q_mv, k_mv, v_mv, q_s, k_s, v_s, basis_q, basis_k, Q, Kf, Vt);

  ga_attn<<<HEADS * (NSEQ / 64), 128, 0, stream>>>(Q, Kf, Vt, (float*)d_out);
}
